// SelfAttention_25529285608073
// MI455X (gfx1250) — matrix operand order repaired and checked by reading, not compiled, not run on hardware
//
#include <hip/hip_runtime.h>

// ---------------------------------------------------------------------------
// Self-attention forward for MI455X (gfx1250, wave32, WMMA, TDM).
// f32->bf16 convert/transpose, TDM-staged QKV GEMM (bf16 WMMA, f32 acc),
// flash attention (bf16 WMMA + online softmax), TDM-staged out-proj GEMM.
// ---------------------------------------------------------------------------

#define BATCH    4
#define SEQ      2048
#define DIM      1024
#define HEADS    16
#define DH       64
#define INNER    1024
#define QKV3     3072
#define MROWS    (BATCH * SEQ)      // 8192
#define SCALE    0.125f             // 64^-0.5

typedef __attribute__((ext_vector_type(16))) __bf16    v16bf;
typedef __attribute__((ext_vector_type(8)))  float     v8f;
typedef __attribute__((ext_vector_type(4)))  unsigned  v4u;
typedef __attribute__((ext_vector_type(8)))  int       v8i_t;
typedef __attribute__((ext_vector_type(4)))  int       v4i_t;

#if defined(__has_builtin)
#if __has_builtin(__builtin_amdgcn_tensor_load_to_lds)
#define HAVE_TDM 1
#endif
#endif
#ifndef HAVE_TDM
#define HAVE_TDM 0
#endif

union FragU {
    v16bf v;
    uint4 q[2];
    unsigned short s[16];
};

__device__ __forceinline__ unsigned short f2bf(float f) {
    unsigned u = __float_as_uint(f);
    u += 0x7FFFu + ((u >> 16) & 1u);          // round-to-nearest-even
    return (unsigned short)(u >> 16);
}

__device__ __forceinline__ v8f zero8() {
    v8f z = {0.f, 0.f, 0.f, 0.f, 0.f, 0.f, 0.f, 0.f};
    return z;
}

__device__ __forceinline__ v8f wmma_bf16(v16bf a, v16bf b, v8f c) {
    return __builtin_amdgcn_wmma_f32_16x16x32_bf16(false, a, false, b,
                                                   (short)0, c, false, false);
}

// A-operand (16 rows x K=32) from row-major bf16 [*, stride].
__device__ __forceinline__ v16bf load_a(const unsigned short* __restrict__ base,
                                        int stride, int row0, int k0) {
    int lane = threadIdx.x & 31;
    int m    = lane & 15;
    int kb   = (lane >> 4) << 3;              // 0 or 8
    const unsigned short* p = base + (size_t)(row0 + m) * stride + k0;
    FragU f;
    f.q[0] = *(const uint4*)(p + kb);
    f.q[1] = *(const uint4*)(p + 16 + kb);
    return f.v;
}

// B-operand (K=32 x 16 cols) where element [k][n] = base[(col0+n)*stride + k0 + k].
__device__ __forceinline__ v16bf load_b(const unsigned short* __restrict__ base,
                                        int stride, int col0, int k0) {
    int lane = threadIdx.x & 31;
    int n    = lane & 15;
    int kb   = (lane >> 4) << 3;              // 0 or 8
    const unsigned short* p = base + (size_t)(col0 + n) * stride + k0;
    FragU f;
    f.q[0] = *(const uint4*)(p + kb);
    f.q[1] = *(const uint4*)(p + 16 + kb);
    return f.v;
}

// LDS tile fragment readers (tiles stored row-major [rows][32] bf16).
// Callers pass static GEPs off __shared__ arrays so these lower to ds_load_b128.
__device__ __forceinline__ v16bf lds_frag_a(const unsigned short* base, int row0) {
    int lane = threadIdx.x & 31;
    int m    = lane & 15;
    int kb   = (lane >> 4) << 3;
    const unsigned short* p = base + (row0 + m) * 32;
    FragU f;
    f.q[0] = *(const uint4*)(p + kb);
    f.q[1] = *(const uint4*)(p + 16 + kb);
    return f.v;
}
__device__ __forceinline__ v16bf lds_frag_b(const unsigned short* base, int col0) {
    int lane = threadIdx.x & 31;
    int n    = lane & 15;
    int kb   = (lane >> 4) << 3;
    const unsigned short* p = base + (col0 + n) * 32;
    FragU f;
    f.q[0] = *(const uint4*)(p + kb);
    f.q[1] = *(const uint4*)(p + 16 + kb);
    return f.v;
}

// ---------------------------------------------------------------------------
// Tensor Data Mover: 2D bf16 tile into LDS (packed [tile_d1][tile_d0]).
#if HAVE_TDM
__device__ __forceinline__ void tdm_load_2d(unsigned lds_off, const void* gaddr,
                                            unsigned tensor_d0, unsigned tensor_d1,
                                            unsigned tile_d0, unsigned tile_d1,
                                            unsigned stride0) {
    unsigned long long ga = (unsigned long long)(size_t)gaddr;
    v4u g0;
    g0[0] = 1u;                                    // count=1, user descriptor
    g0[1] = lds_off;                               // LDS byte address
    g0[2] = (unsigned)(ga & 0xFFFFFFFFu);          // global_addr[31:0]
    g0[3] = (unsigned)((ga >> 32) & 0x1FFFFFFu)    // global_addr[56:32]
          | (2u << 30);                            // type=2 ("image")
    v8i_t g1;
    g1[0] = (int)(1u << 16);                       // data_size=1 -> 2 bytes
    g1[1] = (int)((tensor_d0 & 0xFFFFu) << 16);    // tensor_dim0[15:0]
    g1[2] = (int)((tensor_d0 >> 16) | ((tensor_d1 & 0xFFFFu) << 16));
    g1[3] = (int)((tensor_d1 >> 16) | (tile_d0 << 16));
    g1[4] = (int)(tile_d1 & 0xFFFFu);              // tile_dim1; tile_dim2=0
    g1[5] = (int)stride0;                          // tensor_dim0_stride[31:0]
    g1[6] = 0;
    g1[7] = 0;
    v4i_t z4 = {0, 0, 0, 0};                       // groups 2/3 unused (2D)
    v8i_t z8 = {0, 0, 0, 0, 0, 0, 0, 0};           // extra group (6-arg form)
    __builtin_amdgcn_tensor_load_to_lds(g0, g1, z4, z4, z8, 0);
}
#else
__device__ __forceinline__ void stage_copy(unsigned short* dst,
                                           const unsigned short* src,
                                           int rows, int srcStride) {
    int chunks = rows * 4;                         // 4 uint4 per 32-elem row
    for (int c = threadIdx.x; c < chunks; c += 256) {
        int rr = c >> 2, cc = (c & 3) << 3;
        *(uint4*)(dst + rr * 32 + cc) =
            *(const uint4*)(src + (size_t)rr * srcStride + cc);
    }
}
#endif

// ---------------------------------------------------------------------------
// One K=32 chunk of the 64x64-per-wave compute, operands in LDS.
__device__ __forceinline__ void gemm_compute_chunk(const unsigned short* pa,
                                                   const unsigned short* pb,
                                                   int wr, int wc,
                                                   v8f (&acc)[4][4]) {
    v16bf af[4], bfr[4];
#pragma unroll
    for (int i = 0; i < 4; ++i) af[i]  = lds_frag_a(pa, wr * 64 + 16 * i);
#pragma unroll
    for (int j = 0; j < 4; ++j) bfr[j] = lds_frag_b(pb, wc * 64 + 16 * j);
#pragma unroll
    for (int i = 0; i < 4; ++i)
#pragma unroll
        for (int j = 0; j < 4; ++j)
            acc[i][j] = wmma_bf16(af[i], bfr[j], acc[i][j]);
}

// GEMM core: block tile 128x256, 8 waves of 64x64, K chunks of 32, statically
// double-buffered in LDS via TDM (even chunk -> buf0, odd chunk -> buf1).
__device__ __forceinline__ void gemm_tiles(const unsigned short* __restrict__ Amat,
                                           const unsigned short* __restrict__ Bt,
                                           int nBrows,
                                           unsigned short (&lA)[2][128 * 32],
                                           unsigned short (&lB)[2][256 * 32],
                                           int rowB, int colB,
                                           v8f (&acc)[4][4]) {
    int wave = threadIdx.x >> 5;
    int wr = wave >> 2, wc = wave & 3;
    const unsigned short* Abase = Amat + (size_t)rowB * DIM;
    const unsigned short* Bbase = Bt   + (size_t)colB * DIM;
#if HAVE_TDM
    unsigned offA0 = (unsigned)(size_t)&lA[0][0];
    unsigned offA1 = (unsigned)(size_t)&lA[1][0];
    unsigned offB0 = (unsigned)(size_t)&lB[0][0];
    unsigned offB1 = (unsigned)(size_t)&lB[1][0];

    if (wave == 0) {
        tdm_load_2d(offA0, Abase, DIM, MROWS, 32, 128, DIM);
        tdm_load_2d(offB0, Bbase, DIM, (unsigned)nBrows, 32, 256, DIM);
        __builtin_amdgcn_s_wait_tensorcnt(0);
    }
#else
    stage_copy(&lA[0][0], Abase, 128, DIM);
    stage_copy(&lB[0][0], Bbase, 256, DIM);
#endif
    __syncthreads();

    for (int k0 = 0; k0 < DIM; k0 += 64) {
        // stage odd chunk (k0+32) into buffer 1 while computing buffer 0
#if HAVE_TDM
        if (wave == 0) {
            tdm_load_2d(offA1, Abase + k0 + 32, DIM, MROWS, 32, 128, DIM);
            tdm_load_2d(offB1, Bbase + k0 + 32, DIM, (unsigned)nBrows, 32, 256, DIM);
        }
#else
        stage_copy(&lA[1][0], Abase + k0 + 32, 128, DIM);
        stage_copy(&lB[1][0], Bbase + k0 + 32, 256, DIM);
#endif
        gemm_compute_chunk(&lA[0][0], &lB[0][0], wr, wc, acc);
#if HAVE_TDM
        if (wave == 0) __builtin_amdgcn_s_wait_tensorcnt(0);
#endif
        __syncthreads();

        // stage next even chunk (k0+64) into buffer 0 while computing buffer 1
        bool more = (k0 + 64) < DIM;
        if (more) {
#if HAVE_TDM
            if (wave == 0) {
                tdm_load_2d(offA0, Abase + k0 + 64, DIM, MROWS, 32, 128, DIM);
                tdm_load_2d(offB0, Bbase + k0 + 64, DIM, (unsigned)nBrows, 32, 256, DIM);
            }
#else
            stage_copy(&lA[0][0], Abase + k0 + 64, 128, DIM);
            stage_copy(&lB[0][0], Bbase + k0 + 64, 256, DIM);
#endif
        }
        gemm_compute_chunk(&lA[1][0], &lB[1][0], wr, wc, acc);
#if HAVE_TDM
        if (more && wave == 0) __builtin_amdgcn_s_wait_tensorcnt(0);
#endif
        __syncthreads();
    }
}

// ---------------------------------------------------------------------------
__global__ __launch_bounds__(256) void cvt_bf16(const float* __restrict__ in,
                                                unsigned short* __restrict__ out,
                                                int n) {
    int i = blockIdx.x * 256 + threadIdx.x;
    if (i < n) out[i] = f2bf(in[i]);
}

// out[n*rows + k] = in[k*cols + n]
__global__ __launch_bounds__(256) void transpose_bf16(const float* __restrict__ in,
                                                      unsigned short* __restrict__ out,
                                                      int rows, int cols) {
    int i = blockIdx.x * 256 + threadIdx.x;
    if (i < rows * cols) {
        int n = i / rows;
        int k = i - n * rows;
        out[i] = f2bf(in[(size_t)k * cols + n]);
    }
}

// ---------------------------------------------------------------------------
// QKV GEMM: [8192x1024] x [1024x3072] -> Q,K (B,H,N,dh) and V^T (B,H,dh,N)
__global__ __launch_bounds__(256) void gemm_qkv(
        const unsigned short* __restrict__ xb,
        const unsigned short* __restrict__ wT,   // [3072][1024]
        unsigned short* __restrict__ qb,
        unsigned short* __restrict__ kbuf,
        unsigned short* __restrict__ vT) {
    __shared__ __align__(16) unsigned short lA[2][128 * 32];
    __shared__ __align__(16) unsigned short lB[2][256 * 32];

    v8f acc[4][4];
#pragma unroll
    for (int i = 0; i < 4; ++i)
#pragma unroll
        for (int j = 0; j < 4; ++j) acc[i][j] = zero8();

    int rowB = blockIdx.x * 128;
    int colB = blockIdx.y * 256;
    gemm_tiles(xb, wT, QKV3, lA, lB, rowB, colB, acc);

    int wave = threadIdx.x >> 5;
    int lane = threadIdx.x & 31;
    int rowW = rowB + (wave >> 2) * 64;
    int colW = colB + (wave & 3) * 64;
    int nn   = lane & 15;
    int mofs = (lane >> 4) << 3;
#pragma unroll
    for (int ti = 0; ti < 4; ++ti) {
#pragma unroll
        for (int tj = 0; tj < 4; ++tj) {
            int col   = colW + 16 * tj + nn;
            int which = col >> 10;            // 0=q 1=k 2=v
            int cc    = col & 1023;
            int h     = cc >> 6;
            int d     = cc & 63;
#pragma unroll
            for (int r = 0; r < 8; ++r) {
                int grow = rowW + 16 * ti + mofs + r;
                int bb   = grow >> 11;
                int n    = grow & 2047;
                unsigned short val = f2bf(acc[ti][tj][r]);
                size_t bh = (size_t)bb * HEADS + h;
                if (which == 0)
                    qb[(bh * SEQ + n) * DH + d] = val;
                else if (which == 1)
                    kbuf[(bh * SEQ + n) * DH + d] = val;
                else
                    vT[(bh * DH + d) * SEQ + n] = val;
            }
        }
    }
}

// ---------------------------------------------------------------------------
// Flash attention: one wave per (b,h, 32-query tile). Keys processed 32/iter.
__global__ __launch_bounds__(32) void attn_flash(
        const unsigned short* __restrict__ qb,
        const unsigned short* __restrict__ kbuf,
        const unsigned short* __restrict__ vT,
        unsigned short* __restrict__ obuf) {   // bf16 [8192][1024]
    __shared__ __align__(16) unsigned short pS[32 * 32];

    int lane = threadIdx.x & 31;
    int q0   = blockIdx.x * 32;
    int bh   = blockIdx.y;
    const unsigned short* qp = qb   + (size_t)bh * SEQ * DH;
    const unsigned short* kp = kbuf + (size_t)bh * SEQ * DH;
    const unsigned short* vp = vT   + (size_t)bh * DH * SEQ;

    v16bf qf[2][2];
#pragma unroll
    for (int i = 0; i < 2; ++i)
#pragma unroll
        for (int j = 0; j < 2; ++j)
            qf[i][j] = load_a(qp, DH, q0 + 16 * i, 32 * j);

    float mrow[2][8], lrow[2][8];
    v8f oacc[2][4];
#pragma unroll
    for (int i = 0; i < 2; ++i)
#pragma unroll
        for (int r = 0; r < 8; ++r) { mrow[i][r] = -3.0e38f; lrow[i][r] = 0.f; }
#pragma unroll
    for (int i = 0; i < 2; ++i)
#pragma unroll
        for (int t = 0; t < 4; ++t) oacc[i][t] = zero8();

    int rbase = (lane >> 4) << 3;
    int colp  = lane & 15;

    for (int kc = 0; kc < SEQ; kc += 32) {
        if (kc + 32 < SEQ) {
            __builtin_prefetch(kp + (size_t)(kc + 32) * DH, 0, 0);
            __builtin_prefetch(vp + (size_t)(kc + 32), 0, 0);
        }
        v16bf bk00 = load_b(kp, DH, kc,      0);
        v16bf bk01 = load_b(kp, DH, kc,      32);
        v16bf bk10 = load_b(kp, DH, kc + 16, 0);
        v16bf bk11 = load_b(kp, DH, kc + 16, 32);
        v8f s[2][2];
#pragma unroll
        for (int i = 0; i < 2; ++i) {
            s[i][0] = wmma_bf16(qf[i][1], bk01, wmma_bf16(qf[i][0], bk00, zero8()));
            s[i][1] = wmma_bf16(qf[i][1], bk11, wmma_bf16(qf[i][0], bk10, zero8()));
        }
#pragma unroll
        for (int i = 0; i < 2; ++i) {
            float pa[8], pb[8], alpha[8];
#pragma unroll
            for (int r = 0; r < 8; ++r) {
                float sa = s[i][0][r] * SCALE;
                float sb = s[i][1][r] * SCALE;
                float tm = fmaxf(sa, sb);
                tm = fmaxf(tm, __shfl_xor(tm, 1));
                tm = fmaxf(tm, __shfl_xor(tm, 2));
                tm = fmaxf(tm, __shfl_xor(tm, 4));
                tm = fmaxf(tm, __shfl_xor(tm, 8));
                float mn = fmaxf(mrow[i][r], tm);
                float al = __expf(mrow[i][r] - mn);
                float ea = __expf(sa - mn);
                float eb = __expf(sb - mn);
                float rs = ea + eb;
                rs += __shfl_xor(rs, 1);
                rs += __shfl_xor(rs, 2);
                rs += __shfl_xor(rs, 4);
                rs += __shfl_xor(rs, 8);
                lrow[i][r] = lrow[i][r] * al + rs;
                mrow[i][r] = mn;
                alpha[r] = al; pa[r] = ea; pb[r] = eb;
            }
#pragma unroll
            for (int t = 0; t < 4; ++t)
#pragma unroll
                for (int r = 0; r < 8; ++r)
                    oacc[i][t][r] *= alpha[r];
#pragma unroll
            for (int r = 0; r < 8; ++r) {
                pS[(16 * i + rbase + r) * 32 + colp]      = f2bf(pa[r]);
                pS[(16 * i + rbase + r) * 32 + 16 + colp] = f2bf(pb[r]);
            }
        }
        __syncthreads();
        v16bf pf0 = load_a(pS, 32, 0,  0);
        v16bf pf1 = load_a(pS, 32, 16, 0);
        __syncthreads();
#pragma unroll
        for (int t = 0; t < 4; ++t) {
            v16bf vf = load_b(vp, SEQ, 16 * t, kc);
            oacc[0][t] = wmma_bf16(pf0, vf, oacc[0][t]);
            oacc[1][t] = wmma_bf16(pf1, vf, oacc[1][t]);
        }
    }

    int bb   = bh >> 4;
    int h    = bh & 15;
    int mofs = (lane >> 4) << 3;
#pragma unroll
    for (int i = 0; i < 2; ++i)
#pragma unroll
        for (int t = 0; t < 4; ++t)
#pragma unroll
            for (int r = 0; r < 8; ++r) {
                float v = oacc[i][t][r] / lrow[i][r];
                int n   = q0 + 16 * i + mofs + r;
                int col = h * DH + 16 * t + (lane & 15);
                obuf[((size_t)bb * SEQ + n) * INNER + col] = f2bf(v);
            }
}

// ---------------------------------------------------------------------------
// Out projection: [8192x1024] x [1024x1024] + bias -> f32
__global__ __launch_bounds__(256) void gemm_out(
        const unsigned short* __restrict__ ob,
        const unsigned short* __restrict__ wT,   // [1024][1024]
        const float* __restrict__ bout,
        float* __restrict__ out) {
    __shared__ __align__(16) unsigned short lA[2][128 * 32];
    __shared__ __align__(16) unsigned short lB[2][256 * 32];

    v8f acc[4][4];
#pragma unroll
    for (int i = 0; i < 4; ++i)
#pragma unroll
        for (int j = 0; j < 4; ++j) acc[i][j] = zero8();

    int rowB = blockIdx.x * 128;
    int colB = blockIdx.y * 256;
    gemm_tiles(ob, wT, INNER, lA, lB, rowB, colB, acc);

    int wave = threadIdx.x >> 5;
    int lane = threadIdx.x & 31;
    int rowW = rowB + (wave >> 2) * 64;
    int colW = colB + (wave & 3) * 64;
    int nn   = lane & 15;
    int mofs = (lane >> 4) << 3;
#pragma unroll
    for (int ti = 0; ti < 4; ++ti) {
#pragma unroll
        for (int tj = 0; tj < 4; ++tj) {
            int col = colW + 16 * tj + nn;
            float bias = bout[col];
#pragma unroll
            for (int r = 0; r < 8; ++r) {
                int grow = rowW + 16 * ti + mofs + r;
                out[(size_t)grow * DIM + col] = acc[ti][tj][r] + bias;
            }
        }
    }
}

// ---------------------------------------------------------------------------
extern "C" void kernel_launch(void* const* d_in, const int* in_sizes, int n_in,
                              void* d_out, int out_size, void* d_ws, size_t ws_size,
                              hipStream_t stream) {
    const float* x     = (const float*)d_in[0];
    const float* w_qkv = (const float*)d_in[1];
    const float* w_out = (const float*)d_in[2];
    const float* b_out = (const float*)d_in[3];
    float* out = (float*)d_out;

    char* ws = (char*)d_ws;
    size_t off = 0;
    auto carve = [&](size_t bytes) {
        char* p = ws + off;
        off += (bytes + 255) & ~(size_t)255;
        return p;
    };
    unsigned short* xb    = (unsigned short*)carve((size_t)MROWS * DIM * 2);
    unsigned short* wqkvT = (unsigned short*)carve((size_t)QKV3 * DIM * 2);
    unsigned short* woutT = (unsigned short*)carve((size_t)INNER * DIM * 2);
    unsigned short* qb    = (unsigned short*)carve((size_t)BATCH * HEADS * SEQ * DH * 2);
    unsigned short* kbuf  = (unsigned short*)carve((size_t)BATCH * HEADS * SEQ * DH * 2);
    unsigned short* vT    = (unsigned short*)carve((size_t)BATCH * HEADS * DH * SEQ * 2);
    unsigned short* obuf  = (unsigned short*)carve((size_t)MROWS * INNER * 2);
    (void)ws_size; (void)in_sizes; (void)n_in; (void)out_size;

    {
        int n = MROWS * DIM;
        cvt_bf16<<<(n + 255) / 256, 256, 0, stream>>>(x, xb, n);
    }
    {
        int n = DIM * QKV3;
        transpose_bf16<<<(n + 255) / 256, 256, 0, stream>>>(w_qkv, wqkvT, DIM, QKV3);
    }
    {
        int n = DIM * INNER;
        transpose_bf16<<<(n + 255) / 256, 256, 0, stream>>>(w_out, woutT, DIM, INNER);
    }

    gemm_qkv<<<dim3(MROWS / 128, QKV3 / 256), 256, 0, stream>>>(xb, wqkvT, qb, kbuf, vT);
    attn_flash<<<dim3(SEQ / 32, BATCH * HEADS), 32, 0, stream>>>(qb, kbuf, vT, obuf);
    gemm_out<<<dim3(MROWS / 128, DIM / 256), 256, 0, stream>>>(obuf, woutT, b_out, out);
}
